// Decoder_25881472925937
// MI455X (gfx1250) — matrix operand order repaired and checked by reading, not compiled, not run on hardware
//
#include <hip/hip_runtime.h>
#include <math.h>

#define B_   64
#define P_   196
#define ENC_ 2048
#define H_   512
#define E_   512
#define A_   512
#define V_   30000
#define T_   20
#define TS   19            // T-1 recurrent steps
#define KCAT 3072          // E + ENC + H (fused LSTM input)

typedef __attribute__((ext_vector_type(16))) __bf16 v16bf;
typedef __attribute__((ext_vector_type(8)))  __bf16 v8bf;
typedef __attribute__((ext_vector_type(8)))  float  v8f;
typedef __attribute__((ext_vector_type(4)))  int    v4i;

__device__ __forceinline__ float sigmoidf_(float x) { return 1.0f / (1.0f + __expf(-x)); }

// ---- CDNA5 async global->LDS copy (ASYNCcnt path), builtin with asm fallback ----
__device__ __forceinline__ void async_copy_b128(const void* g, void* l)
{
#if __has_builtin(__builtin_amdgcn_global_load_async_to_lds_b128)
    typedef __attribute__((address_space(1))) v4i* gptr_t;   // global v4i*
    typedef __attribute__((address_space(3))) v4i* lptr_t;   // LDS v4i*
    __builtin_amdgcn_global_load_async_to_lds_b128((gptr_t)(void*)g, (lptr_t)l, 0, 0);
#else
    unsigned l32 = (unsigned)(unsigned long long)l;
    asm volatile("global_load_async_to_lds_b128 %0, %1, off"
                 :: "v"(l32), "v"(g) : "memory");
#endif
}

__device__ __forceinline__ void wait_async0()
{
#if __has_builtin(__builtin_amdgcn_s_wait_asynccnt)
    __builtin_amdgcn_s_wait_asynccnt(0);
#else
    asm volatile("s_wait_asynccnt 0x0" ::: "memory");
#endif
}

// ---------------------------------------------------------------------------
// WMMA GEMM: C[M,N](f32) = A[M,K](bf16,row-major) @ W[N,K](bf16)^T + bias
// Block: 256 threads = 8 waves; block tile 128(M) x 64(N); wave tile 16 x 64.
// B tile (64 rows x 32 K, shared by all 8 waves) staged through LDS with
// double-buffered async global->LDS copies; A fragments double-buffered in VGPRs.
// All 8 ds_load_b128 for the 4 B fragments are issued before the WMMA chain so
// dscnt waits graduate instead of serializing each WMMA on its own loads.
// out_mode 0: C[m*ldc + n]
// out_mode 1: (vocab) t=m>>6, b=m&63 -> C[b*strideB + t*strideT + n]
// ---------------------------------------------------------------------------
__global__ __launch_bounds__(256)
void gemm_bf16_wmma_kernel(const __bf16* __restrict__ A, int lda,
                           const __bf16* __restrict__ W, int ldw,
                           const float* __restrict__ bias,
                           float* __restrict__ C, long long ldc_or_strideB, int strideT,
                           int M, int N, int K, int out_mode)
{
    __shared__ __align__(32) __bf16 Bs[2][64][32];   // 8 KB double-buffered B tile

    const int lane = threadIdx.x & 31;
    const int wave = threadIdx.x >> 5;
    int m0 = blockIdx.y * 128 + wave * 16;
    if (m0 >= M) m0 = M - 16;          // clamp (dup work, keeps barriers convergent)
    const int n0 = blockIdx.x * 64;

    // async fill of one 64x32 B tile: 256 threads x 16B segments
    auto fill_tile = [&](int buf, int k0f) {
        const int row = threadIdx.x >> 2;     // 0..63  (local n)
        const int seg = threadIdx.x & 3;      // 4 x 16B per row
        int nRow = n0 + row;
        if (nRow >= N) nRow = N - 1;          // clamp ragged N: loads stay valid
        const __bf16* gp = W + (long long)nRow * ldw + k0f + seg * 8;
        async_copy_b128(gp, &Bs[buf][row][seg * 8]);
    };

    v8f acc[4];
#pragma unroll
    for (int j = 0; j < 4; ++j)
#pragma unroll
        for (int e = 0; e < 8; ++e) acc[j][e] = 0.0f;

    // A fragment addressing (16-bit A 16x32 layout): lane<16 -> K {0..7,16..23},
    // lane>=16 -> K {8..15,24..31}; row = m0 + (lane&15)
    const int aRow = m0 + (lane & 15);
    const int kA   = (lane < 16) ? 0 : 8;
    const int kB   = (lane < 16) ? 0 : 8;    // B frag K-half selector in LDS tile
    const __bf16* apBase = A + (long long)aRow * lda + kA;

    union AF { v16bf v; v8bf h[2]; };
    AF a_cur, a_nxt;
    a_cur.h[0] = *(const v8bf*)(apBase);
    a_cur.h[1] = *(const v8bf*)(apBase + 16);

    fill_tile(0, 0);                        // prologue fill buf 0
    int cur = 0;

    for (int k0 = 0; k0 < K; k0 += 32) {
        wait_async0();                      // my fills into buf `cur` done
        __syncthreads();                    // everyone's fills done; prev reads of 1-cur done

        const bool more = (k0 + 32) < K;
        if (more) {
            fill_tile(1 - cur, k0 + 32);    // overlap next B tile with this slab's WMMAs
            const __bf16* ap = apBase + k0 + 32;
            a_nxt.h[0] = *(const v8bf*)(ap);
            a_nxt.h[1] = *(const v8bf*)(ap + 16);
        }

        // issue ALL B-fragment LDS reads first (clause), then the WMMA chain
        AF bfr[4];
#pragma unroll
        for (int j = 0; j < 4; ++j) {
            const int nl = (lane & 15) + j * 16;
            bfr[j].h[0] = *(const v8bf*)&Bs[cur][nl][kB];       // ds_load_b128 x2
            bfr[j].h[1] = *(const v8bf*)&Bs[cur][nl][kB + 16];
        }
#pragma unroll
        for (int j = 0; j < 4; ++j) {
            acc[j] = __builtin_amdgcn_wmma_f32_16x16x32_bf16(
                false, a_cur.v, false, bfr[j].v, (short)0, acc[j], false, false);
        }

        if (more) a_cur = a_nxt;
        cur ^= 1;
    }

    // Epilogue: C/D layout: VGPR r, lanes 0-15 -> M=m0+r, lanes 16-31 -> M=m0+r+8
    const int mOff = (lane < 16) ? 0 : 8;
    const int nCol = lane & 15;
#pragma unroll
    for (int j = 0; j < 4; ++j) {
        const int n = n0 + j * 16 + nCol;
        if (n >= N) continue;               // wave-uniform (N multiple of 16)
        const float bv = bias ? bias[n] : 0.0f;
#pragma unroll
        for (int r = 0; r < 8; ++r) {
            const int m = m0 + r + mOff;
            const float val = acc[j][r] + bv;
            if (out_mode == 0) {
                C[(long long)m * ldc_or_strideB + n] = val;
            } else {
                const int t = m >> 6, b = m & 63;
                C[(long long)b * ldc_or_strideB + (long long)t * strideT + n] = val;
            }
        }
    }
}

// ---------------------------------------------------------------------------
// Elementwise / setup kernels
// ---------------------------------------------------------------------------
__global__ void cvt_kernel(const float* __restrict__ src, __bf16* __restrict__ dst, long long n)
{
    long long i = (long long)blockIdx.x * blockDim.x + threadIdx.x;
    const long long stride = (long long)gridDim.x * blockDim.x;
    for (; i < n; i += stride) dst[i] = (__bf16)src[i];
}

__global__ void wcat_kernel(const float* __restrict__ W_ih, const float* __restrict__ W_hh,
                            __bf16* __restrict__ Wcat)
{
    long long idx = (long long)blockIdx.x * blockDim.x + threadIdx.x;
    const long long n = (long long)(4 * H_) * KCAT;
    if (idx >= n) return;
    const int j = (int)(idx % KCAT);
    const int r = (int)(idx / KCAT);
    const float v = (j < E_ + ENC_) ? W_ih[(long long)r * (E_ + ENC_) + j]
                                    : W_hh[(long long)r * H_ + (j - (E_ + ENC_))];
    Wcat[idx] = (__bf16)v;
}

__global__ void bsum_kernel(const float* __restrict__ b_ih, const float* __restrict__ b_hh,
                            float* __restrict__ bsum)
{
    const int i = blockIdx.x * blockDim.x + threadIdx.x;
    if (i < 4 * H_) bsum[i] = b_ih[i] + b_hh[i];
}

__global__ void embed_kernel(const int* __restrict__ captions, const float* __restrict__ emb,
                             __bf16* __restrict__ embeds)
{
    const long long idx = (long long)blockIdx.x * blockDim.x + threadIdx.x;
    if (idx >= (long long)TS * B_ * E_) return;
    const int e = (int)(idx & (E_ - 1));
    const int b = (int)((idx >> 9) & 63);
    const int t = (int)(idx >> 15);
    const int tok = captions[b * T_ + t];
    embeds[idx] = (__bf16)emb[(long long)tok * E_ + e];
}

__global__ void mean_kernel(const float* __restrict__ enc, __bf16* __restrict__ mean_bf)
{
    const int b = blockIdx.x;
    const int e = blockIdx.y * 256 + threadIdx.x;
    const float* base = enc + ((long long)b * P_) * ENC_ + e;
    float s = 0.0f;
    for (int p = 0; p < P_; ++p) s += base[(long long)p * ENC_];
    mean_bf[b * ENC_ + e] = (__bf16)(s * (1.0f / (float)P_));
}

__global__ void preph_kernel(const float* __restrict__ h, __bf16* __restrict__ inp,
                             __bf16* __restrict__ sigh)
{
    const int idx = blockIdx.x * blockDim.x + threadIdx.x;
    if (idx >= B_ * H_) return;
    const int b = idx >> 9, j = idx & 511;
    const float v = h[idx];
    inp[(long long)b * KCAT + (E_ + ENC_) + j] = (__bf16)v;   // h_t -> last K slice
    sigh[idx] = (__bf16)sigmoidf_(v);                          // sigmoid(h) for gate GEMM
}

__global__ void lstm_kernel(const float* __restrict__ gates, float* __restrict__ h,
                            float* __restrict__ c, __bf16* __restrict__ hist_row)
{
    const int idx = blockIdx.x * blockDim.x + threadIdx.x;
    if (idx >= B_ * H_) return;
    const int b = idx >> 9, j = idx & 511;
    const float* g = gates + (long long)b * (4 * H_);
    const float ig = sigmoidf_(g[j]);
    const float fg = sigmoidf_(g[H_ + j]);
    const float gg = tanhf(g[2 * H_ + j]);
    const float og = sigmoidf_(g[3 * H_ + j]);
    const float cn = fg * c[idx] + ig * gg;
    const float hn = og * tanhf(cn);
    c[idx] = cn;
    h[idx] = hn;
    hist_row[(long long)b * H_ + j] = (__bf16)hn;
}

__global__ void zerotail_kernel(float* __restrict__ out)
{
    const int idx = blockIdx.x * blockDim.x + threadIdx.x;
    if (idx >= B_ * V_) return;
    const int b = idx / V_, v = idx % V_;
    out[(long long)b * T_ * V_ + (long long)(T_ - 1) * V_ + v] = 0.0f;
}

// ---------------------------------------------------------------------------
// Attention: scores -> softmax over P -> weighted encoder sum * gate -> inp
// One block per batch element b.
// ---------------------------------------------------------------------------
__global__ __launch_bounds__(256)
void attention_kernel(const float* __restrict__ att_enc, const float* __restrict__ att_h,
                      const float* __restrict__ Wa, const float* __restrict__ ba,
                      const float* __restrict__ enc, const float* __restrict__ gate,
                      const __bf16* __restrict__ emb_t, __bf16* __restrict__ inp)
{
    __shared__ float sh_atth[A_];
    __shared__ float sh_alpha[P_];
    __shared__ float red[16];
    const int b = blockIdx.x;
    const int tid = threadIdx.x;
    const int lane = tid & 31, wave = tid >> 5;

    for (int a = tid; a < A_; a += 256) sh_atth[a] = att_h[b * A_ + a];
    __syncthreads();

    float myscore = -3.0e38f;
    if (tid < P_) {
        const float* ae = att_enc + ((long long)b * P_ + tid) * A_;
        float s = ba[0];
        for (int a = 0; a < A_; a += 4) {
            const float4 v = *(const float4*)(ae + a);
            const float4 w = *(const float4*)(Wa + a);
            s += w.x * fmaxf(v.x + sh_atth[a + 0], 0.0f);
            s += w.y * fmaxf(v.y + sh_atth[a + 1], 0.0f);
            s += w.z * fmaxf(v.z + sh_atth[a + 2], 0.0f);
            s += w.w * fmaxf(v.w + sh_atth[a + 3], 0.0f);
        }
        myscore = s;
    }
    float wm = myscore;
    for (int o = 16; o > 0; o >>= 1) wm = fmaxf(wm, __shfl_xor(wm, o, 32));
    if (lane == 0) red[wave] = wm;
    __syncthreads();
    if (tid == 0) { float m = red[0]; for (int i = 1; i < 8; ++i) m = fmaxf(m, red[i]); red[8] = m; }
    __syncthreads();
    const float mx = red[8];
    float ev = (tid < P_) ? __expf(myscore - mx) : 0.0f;
    float ss = ev;
    for (int o = 16; o > 0; o >>= 1) ss += __shfl_xor(ss, o, 32);
    if (lane == 0) red[wave] = ss;
    __syncthreads();
    if (tid == 0) { float s2 = 0.0f; for (int i = 0; i < 8; ++i) s2 += red[i]; red[9] = 1.0f / s2; }
    __syncthreads();
    if (tid < P_) sh_alpha[tid] = ev * red[9];
    __syncthreads();

    const int e0 = tid * 8;
    float acc[8];
#pragma unroll
    for (int q = 0; q < 8; ++q) acc[q] = 0.0f;
    const float* ebase = enc + (long long)b * P_ * ENC_ + e0;
    for (int p = 0; p < P_; ++p) {
        const float al = sh_alpha[p];
        const float4 u = *(const float4*)(ebase + (long long)p * ENC_);
        const float4 v = *(const float4*)(ebase + (long long)p * ENC_ + 4);
        acc[0] += al * u.x; acc[1] += al * u.y; acc[2] += al * u.z; acc[3] += al * u.w;
        acc[4] += al * v.x; acc[5] += al * v.y; acc[6] += al * v.z; acc[7] += al * v.w;
    }
#pragma unroll
    for (int q = 0; q < 8; ++q) {
        const float g = gate[b * ENC_ + e0 + q];
        inp[(long long)b * KCAT + E_ + e0 + q] = (__bf16)(acc[q] * g);
    }
    for (int e2 = tid; e2 < E_; e2 += 256)
        inp[(long long)b * KCAT + e2] = emb_t[b * E_ + e2];
}

// ---------------------------------------------------------------------------
extern "C" void kernel_launch(void* const* d_in, const int* in_sizes, int n_in,
                              void* d_out, int out_size, void* d_ws, size_t ws_size,
                              hipStream_t stream)
{
    (void)in_sizes; (void)n_in; (void)out_size; (void)ws_size;

    const float* encode_out = (const float*)d_in[0];
    const int*   captions   = (const int*)  d_in[1];
    const float* emb   = (const float*)d_in[2];
    const float* W_ih  = (const float*)d_in[3];
    const float* b_ih  = (const float*)d_in[4];
    const float* W_hh  = (const float*)d_in[5];
    const float* b_hh  = (const float*)d_in[6];
    const float* We    = (const float*)d_in[7];
    const float* be    = (const float*)d_in[8];
    const float* Wh    = (const float*)d_in[9];
    const float* bh    = (const float*)d_in[10];
    const float* Wa    = (const float*)d_in[11];
    const float* ba    = (const float*)d_in[12];
    const float* Wch   = (const float*)d_in[13];
    const float* bch   = (const float*)d_in[14];
    const float* Wcc   = (const float*)d_in[15];
    const float* bcc   = (const float*)d_in[16];
    const float* Wfc   = (const float*)d_in[17];
    const float* bfc   = (const float*)d_in[18];
    const float* Wg    = (const float*)d_in[19];
    const float* bg    = (const float*)d_in[20];
    float* out = (float*)d_out;

    char* ws = (char*)d_ws;
    size_t off = 0;
    auto alloc = [&](size_t bytes) -> char* {
        char* p = ws + off;
        off += (bytes + 255) & ~(size_t)255;
        return p;
    };
    __bf16* enc_bf    = (__bf16*)alloc((size_t)B_ * P_ * ENC_ * 2);
    __bf16* We_bf     = (__bf16*)alloc((size_t)A_ * ENC_ * 2);
    __bf16* Wh_bf     = (__bf16*)alloc((size_t)A_ * H_ * 2);
    __bf16* Wg_bf     = (__bf16*)alloc((size_t)ENC_ * H_ * 2);
    __bf16* Wch_bf    = (__bf16*)alloc((size_t)H_ * ENC_ * 2);
    __bf16* Wcc_bf    = (__bf16*)alloc((size_t)H_ * ENC_ * 2);
    __bf16* Wfc_bf    = (__bf16*)alloc((size_t)V_ * H_ * 2);
    __bf16* Wcat_bf   = (__bf16*)alloc((size_t)(4 * H_) * KCAT * 2);
    float*  att_enc   = (float*) alloc((size_t)B_ * P_ * A_ * 4);
    __bf16* mean_bf   = (__bf16*)alloc((size_t)B_ * ENC_ * 2);
    float*  h_st      = (float*) alloc((size_t)B_ * H_ * 4);
    float*  c_st      = (float*) alloc((size_t)B_ * H_ * 4);
    float*  att_h     = (float*) alloc((size_t)B_ * A_ * 4);
    float*  gate      = (float*) alloc((size_t)B_ * ENC_ * 4);
    __bf16* sigh_bf   = (__bf16*)alloc((size_t)B_ * H_ * 2);
    __bf16* inp_bf    = (__bf16*)alloc((size_t)B_ * KCAT * 2);
    float*  gates     = (float*) alloc((size_t)B_ * 4 * H_ * 4);
    __bf16* hist_bf   = (__bf16*)alloc((size_t)TS * B_ * H_ * 2);
    __bf16* embeds_bf = (__bf16*)alloc((size_t)TS * B_ * E_ * 2);
    float*  bsum      = (float*) alloc((size_t)4 * H_ * 4);

    auto cvtN = [&](const float* s, __bf16* d, long long n) {
        int grid = (int)((n + 255) / 256);
        if (grid > 4096) grid = 4096;
        cvt_kernel<<<grid, 256, 0, stream>>>(s, d, n);
    };
    auto launch_gemm = [&](const __bf16* Aa, int lda, const __bf16* Ww, int ldw,
                           const float* bias, float* Cc, long long ldcOrStrideB, int strideT,
                           int M, int N, int K, int mode) {
        dim3 g((N + 63) / 64, (M + 127) / 128);
        gemm_bf16_wmma_kernel<<<g, 256, 0, stream>>>(Aa, lda, Ww, ldw, bias, Cc,
                                                     ldcOrStrideB, strideT, M, N, K, mode);
    };

    // ---- one-time setup ----
    cvtN(encode_out, enc_bf, (long long)B_ * P_ * ENC_);
    cvtN(We,  We_bf,  (long long)A_ * ENC_);
    cvtN(Wh,  Wh_bf,  (long long)A_ * H_);
    cvtN(Wg,  Wg_bf,  (long long)ENC_ * H_);
    cvtN(Wch, Wch_bf, (long long)H_ * ENC_);
    cvtN(Wcc, Wcc_bf, (long long)H_ * ENC_);
    cvtN(Wfc, Wfc_bf, (long long)V_ * H_);
    wcat_kernel<<<(int)(((long long)(4 * H_) * KCAT + 255) / 256), 256, 0, stream>>>(W_ih, W_hh, Wcat_bf);
    bsum_kernel<<<(4 * H_ + 255) / 256, 256, 0, stream>>>(b_ih, b_hh, bsum);
    embed_kernel<<<(int)(((long long)TS * B_ * E_ + 255) / 256), 256, 0, stream>>>(captions, emb, embeds_bf);
    mean_kernel<<<dim3(B_, ENC_ / 256), 256, 0, stream>>>(encode_out, mean_bf);

    // h0 / c0
    launch_gemm(mean_bf, ENC_, Wch_bf, ENC_, bch, h_st, H_, 0, B_, H_, ENC_, 0);
    launch_gemm(mean_bf, ENC_, Wcc_bf, ENC_, bcc, c_st, H_, 0, B_, H_, ENC_, 0);
    // loop-invariant encoder attention projection (big WMMA GEMM)
    launch_gemm(enc_bf, ENC_, We_bf, ENC_, be, att_enc, A_, 0, B_ * P_, A_, ENC_, 0);

    // ---- recurrent scan (only skinny M=64 GEMMs stay inside) ----
    for (int t = 0; t < TS; ++t) {
        preph_kernel<<<(B_ * H_) / 256, 256, 0, stream>>>(h_st, inp_bf, sigh_bf);
        launch_gemm(inp_bf + (E_ + ENC_), KCAT, Wh_bf, H_, bh, att_h, A_, 0, B_, A_, H_, 0);
        launch_gemm(sigh_bf, H_, Wg_bf, H_, bg, gate, ENC_, 0, B_, ENC_, H_, 0);
        attention_kernel<<<B_, 256, 0, stream>>>(att_enc, att_h, Wa, ba, encode_out, gate,
                                                 embeds_bf + (long long)t * B_ * E_, inp_bf);
        launch_gemm(inp_bf, KCAT, Wcat_bf, KCAT, bsum, gates, 4 * H_, 0, B_, 4 * H_, KCAT, 0);
        lstm_kernel<<<(B_ * H_) / 256, 256, 0, stream>>>(gates, h_st, c_st,
                                                         hist_bf + (long long)t * B_ * H_);
    }

    // ---- hoisted vocab projection for all 19 steps at once (dominant GEMM) ----
    launch_gemm(hist_bf, H_, Wfc_bf, H_, bfc, out, (long long)T_ * V_, V_, TS * B_, V_, H_, 1);
    zerotail_kernel<<<(B_ * V_ + 255) / 256, 256, 0, stream>>>(out);
}
